// QuantumFFN_85160611545401
// MI455X (gfx1250) — compile-verified, matrix operand order checked by reading, not run on hardware
//
#include <hip/hip_runtime.h>
#include <hip/hip_bf16.h>

// Complex FFN: up = modrelu(x @ Wup^T + bias); down = up @ Wdown^T;
// out = x + complex_layernorm(down).
// Strategy: bf16 WMMA (v_wmma_f32_16x16x32_bf16) for both complex GEMMs
// (4 real WMMAs per complex tile), f32 accumulation, fused epilogues.
// Pass 2 stages its bf16 A operand with GLOBAL_LOAD_ASYNC_TO_LDS (ASYNCcnt)
// when the toolchain exposes the builtins.

typedef __attribute__((ext_vector_type(16))) __bf16 v16bf;
typedef __attribute__((ext_vector_type(8)))  __bf16 v8bf;
typedef __attribute__((ext_vector_type(8)))  float  v8f;
typedef __attribute__((ext_vector_type(4)))  int    v4i;

#define BM 128
#define BN 64
#define BK 32
#define EPSL 1e-5f

#if defined(__AMDGCN__) && __has_builtin(__builtin_amdgcn_global_load_async_to_lds_b128) && \
    __has_builtin(__builtin_amdgcn_s_wait_asynccnt)
#define USE_ASYNC_LDS 1
typedef __attribute__((address_space(1))) v4i as1_v4i;  // global-AS int4 (builtin param type)
typedef __attribute__((address_space(3))) v4i as3_v4i;  // LDS-AS int4
#else
#define USE_ASYNC_LDS 0
#endif

__device__ inline unsigned pack_bf16(float a, float b) {
    union { __bf16 h[2]; unsigned u; } t;
    t.h[0] = (__bf16)a; t.h[1] = (__bf16)b;
    return t.u;
}

// Negate a bf16 fragment by flipping packed sign bits (bf16 WMMA has no A/B neg).
__device__ inline v16bf neg_frag(v16bf a) {
    union { v16bf v; unsigned u[8]; } t;
    t.v = a;
#pragma unroll
    for (int i = 0; i < 8; i++) t.u[i] ^= 0x80008000u;
    return t.v;
}

// 16-bit A/B fragment (16x32, row-major LDS tile, row stride BK):
// lane<16 holds K {0..7, 16..23} of row (lane&15); lane>=16 holds K {8..15, 24..31}.
__device__ inline v16bf load_frag(const __bf16* sm, int row, int lk) {
    v8bf lo = *(const v8bf*)(sm + row * BK + lk);
    v8bf hi = *(const v8bf*)(sm + row * BK + lk + 16);
    v16bf f;
#pragma unroll
    for (int i = 0; i < 8; i++) { f[i] = lo[i]; f[8 + i] = hi[i]; }
    return f;
}

// A_BF16: A operand is bf16 in memory (pass 2) vs f32 (pass 1, converted on load).
// EPI==0: bias + modReLU epilogue, bf16 outputs. EPI==1: plain f32 store.
template <bool A_BF16, int EPI>
__global__ __launch_bounds__(256) void cgemm_kernel(
    const void* __restrict__ Are_, const void* __restrict__ Aim_,
    const float* __restrict__ Bre, const float* __restrict__ Bim,
    const float* __restrict__ bias_re, const float* __restrict__ bias_im,
    const float* __restrict__ mrelu_b,
    void* __restrict__ Ore_, void* __restrict__ Oim_,
    int M, int N, int K)
{
    __shared__ __align__(16) __bf16 smA_re[2][BM * BK];
    __shared__ __align__(16) __bf16 smA_im[2][BM * BK];
    __shared__ __align__(16) __bf16 smB_re[2][BN * BK];
    __shared__ __align__(16) __bf16 smB_im[2][BN * BK];

    const int tid  = threadIdx.x;
    const int lane = tid & 31;
    const int wave = tid >> 5;
    const int wm   = wave & 3;   // 4 wave-rows x 32
    const int wn   = wave >> 2;  // 2 wave-cols x 32
    const int row0 = blockIdx.y * BM;
    const int col0 = blockIdx.x * BN;
    const int nk   = K / BK;

    v8f acc_re[2][2] = {};
    v8f acc_im[2][2] = {};

    float4 a_re_f[4], a_im_f[4];  // !A_BF16 staging
    uint4  a_re_b[2], a_im_b[2];  // A_BF16 staging (non-async fallback)
    float4 b_re_f[2], b_im_f[2];

    // Async direct global->LDS staging of the bf16 A tile (pass 2 only).
    auto issue_asyncA = [&](int kt, int buf) {
#if USE_ASYNC_LDS
        if (A_BF16) {
            const __bf16* Are = (const __bf16*)Are_;
            const __bf16* Aim = (const __bf16*)Aim_;
            const int k0 = kt * BK;
#pragma unroll
            for (int i = 0; i < 2; i++) {
                int c = tid + i * 256;          // 512 16B chunks (128x32 bf16)
                int r = c >> 2, kk = (c & 3) << 3;
                size_t goff = (size_t)(row0 + r) * K + k0 + kk;
                int    loff = r * BK + kk;
                __builtin_amdgcn_global_load_async_to_lds_b128(
                    (as1_v4i*)(const void*)(Are + goff),
                    (as3_v4i*)(void*)&smA_re[buf][loff], 0, 0);
                __builtin_amdgcn_global_load_async_to_lds_b128(
                    (as1_v4i*)(const void*)(Aim + goff),
                    (as3_v4i*)(void*)&smA_im[buf][loff], 0, 0);
            }
        }
#else
        (void)kt; (void)buf;
#endif
    };

    auto wait_asyncA = [&]() {
#if USE_ASYNC_LDS
        if (A_BF16) __builtin_amdgcn_s_wait_asynccnt(0);
#endif
    };

    auto load_global = [&](int kt) {
        const int k0 = kt * BK;
        if (!A_BF16) {
            const float* Are = (const float*)Are_;
            const float* Aim = (const float*)Aim_;
#pragma unroll
            for (int i = 0; i < 4; i++) {
                int c = tid + i * 256;          // 1024 float4 chunks (128x32)
                int r = c >> 3, kk = (c & 7) << 2;
                a_re_f[i] = *(const float4*)(Are + (size_t)(row0 + r) * K + k0 + kk);
                a_im_f[i] = *(const float4*)(Aim + (size_t)(row0 + r) * K + k0 + kk);
            }
        } else {
#if !USE_ASYNC_LDS
            const __bf16* Are = (const __bf16*)Are_;
            const __bf16* Aim = (const __bf16*)Aim_;
#pragma unroll
            for (int i = 0; i < 2; i++) {
                int c = tid + i * 256;          // 512 8-elem chunks (128x32 bf16)
                int r = c >> 2, kk = (c & 3) << 3;
                a_re_b[i] = *(const uint4*)(Are + (size_t)(row0 + r) * K + k0 + kk);
                a_im_b[i] = *(const uint4*)(Aim + (size_t)(row0 + r) * K + k0 + kk);
            }
#endif
        }
#pragma unroll
        for (int i = 0; i < 2; i++) {
            int c = tid + i * 256;              // 512 float4 chunks (64x32)
            int r = c >> 3, kk = (c & 7) << 2;
            b_re_f[i] = *(const float4*)(Bre + (size_t)(col0 + r) * K + k0 + kk);
            b_im_f[i] = *(const float4*)(Bim + (size_t)(col0 + r) * K + k0 + kk);
        }
    };

    auto store_lds = [&](int buf) {
        if (!A_BF16) {
#pragma unroll
            for (int i = 0; i < 4; i++) {
                int c = tid + i * 256;
                int r = c >> 3, kk = (c & 7) << 2;
                unsigned* p = (unsigned*)&smA_re[buf][r * BK + kk];
                p[0] = pack_bf16(a_re_f[i].x, a_re_f[i].y);
                p[1] = pack_bf16(a_re_f[i].z, a_re_f[i].w);
                unsigned* q = (unsigned*)&smA_im[buf][r * BK + kk];
                q[0] = pack_bf16(a_im_f[i].x, a_im_f[i].y);
                q[1] = pack_bf16(a_im_f[i].z, a_im_f[i].w);
            }
        } else {
#if !USE_ASYNC_LDS
#pragma unroll
            for (int i = 0; i < 2; i++) {
                int c = tid + i * 256;
                int r = c >> 2, kk = (c & 3) << 3;
                *(uint4*)&smA_re[buf][r * BK + kk] = a_re_b[i];
                *(uint4*)&smA_im[buf][r * BK + kk] = a_im_b[i];
            }
#endif
        }
#pragma unroll
        for (int i = 0; i < 2; i++) {
            int c = tid + i * 256;
            int r = c >> 3, kk = (c & 7) << 2;
            unsigned* p = (unsigned*)&smB_re[buf][r * BK + kk];
            p[0] = pack_bf16(b_re_f[i].x, b_re_f[i].y);
            p[1] = pack_bf16(b_re_f[i].z, b_re_f[i].w);
            unsigned* q = (unsigned*)&smB_im[buf][r * BK + kk];
            q[0] = pack_bf16(b_im_f[i].x, b_im_f[i].y);
            q[1] = pack_bf16(b_im_f[i].z, b_im_f[i].w);
        }
    };

    // Prologue: stage tile 0.
    load_global(0);
    issue_asyncA(0, 0);
    store_lds(0);
    wait_asyncA();
    __syncthreads();

    for (int kt = 0; kt < nk; ++kt) {
        const int cur = kt & 1;
        if (kt + 1 < nk) {
            load_global(kt + 1);
            // Safe: buffer cur^1 was last read in iteration kt-1, which ended
            // with a full barrier before this point.
            issue_asyncA(kt + 1, cur ^ 1);
        }
        if (kt + 2 < nk) {
            // L2 prefetch of the B panel two steps ahead (global_prefetch_b8)
            __builtin_prefetch(Bre + (size_t)(col0 + (tid & 63)) * K + (kt + 2) * BK, 0, 0);
        }

        const int lrow = lane & 15;
        const int lk   = (lane >> 4) << 3;

        v16bf a_re[2], a_im[2], a_imn[2], b_re2[2], b_im2[2];
#pragma unroll
        for (int mi = 0; mi < 2; mi++) {
            int r = wm * 32 + mi * 16 + lrow;
            a_re[mi]  = load_frag(smA_re[cur], r, lk);
            a_im[mi]  = load_frag(smA_im[cur], r, lk);
            a_imn[mi] = neg_frag(a_im[mi]);
        }
#pragma unroll
        for (int ni = 0; ni < 2; ni++) {
            int r = wn * 32 + ni * 16 + lrow;
            b_re2[ni] = load_frag(smB_re[cur], r, lk);
            b_im2[ni] = load_frag(smB_im[cur], r, lk);
        }

#pragma unroll
        for (int mi = 0; mi < 2; mi++)
#pragma unroll
            for (int ni = 0; ni < 2; ni++) {
                // Cre += Are*Bre - Aim*Bim ; Cim += Are*Bim + Aim*Bre
                acc_re[mi][ni] = __builtin_amdgcn_wmma_f32_16x16x32_bf16(
                    false, a_re[mi], false, b_re2[ni], (short)0, acc_re[mi][ni], false, false);
                acc_re[mi][ni] = __builtin_amdgcn_wmma_f32_16x16x32_bf16(
                    false, a_imn[mi], false, b_im2[ni], (short)0, acc_re[mi][ni], false, false);
                acc_im[mi][ni] = __builtin_amdgcn_wmma_f32_16x16x32_bf16(
                    false, a_re[mi], false, b_im2[ni], (short)0, acc_im[mi][ni], false, false);
                acc_im[mi][ni] = __builtin_amdgcn_wmma_f32_16x16x32_bf16(
                    false, a_im[mi], false, b_re2[ni], (short)0, acc_im[mi][ni], false, false);
            }

        if (kt + 1 < nk) {
            __syncthreads();
            store_lds(cur ^ 1);
            wait_asyncA();   // own asyncs done before the barrier below
        }
        __syncthreads();
    }

    // Epilogue. C/D layout: VGPR j, lanes 0-15 -> M=j, lanes 16-31 -> M=j+8; N = lane&15.
#pragma unroll
    for (int mi = 0; mi < 2; mi++) {
#pragma unroll
        for (int ni = 0; ni < 2; ni++) {
            const int mbase = row0 + wm * 32 + mi * 16 + ((lane >> 4) << 3);
            const int n     = col0 + wn * 32 + ni * 16 + (lane & 15);
            if (EPI == 0) {
                const float bre = bias_re[n], bim = bias_im[n], mb = mrelu_b[n];
                __bf16* Ore = (__bf16*)Ore_;
                __bf16* Oim = (__bf16*)Oim_;
#pragma unroll
                for (int j = 0; j < 8; j++) {
                    float cr  = acc_re[mi][ni][j] + bre;
                    float ci  = acc_im[mi][ni][j] + bim;
                    float mag = sqrtf(cr * cr + ci * ci);
                    float s   = fmaxf(mag + mb, 0.0f) / (mag + EPSL);
                    size_t off = (size_t)(mbase + j) * N + n;
                    Ore[off] = (__bf16)(cr * s);
                    Oim[off] = (__bf16)(ci * s);
                }
            } else {
                float* Ore = (float*)Ore_;
                float* Oim = (float*)Oim_;
#pragma unroll
                for (int j = 0; j < 8; j++) {
                    size_t off = (size_t)(mbase + j) * N + n;
                    Ore[off] = acc_re[mi][ni][j];
                    Oim[off] = acc_im[mi][ni][j];
                }
            }
        }
    }
}

// Complex layernorm over D + gamma/beta + residual, interleaved complex output.
__global__ __launch_bounds__(256) void ln_residual_kernel(
    const float* __restrict__ dre, const float* __restrict__ dim_,
    const float* __restrict__ xre, const float* __restrict__ xim,
    const float* __restrict__ gre, const float* __restrict__ gim,
    const float* __restrict__ bre, const float* __restrict__ bim,
    float* __restrict__ out, int Dn)
{
    const int row = blockIdx.x;
    const int tid = threadIdx.x;
    const int PER = Dn / 256;  // 8 for D=2048
    const float* r_re = dre + (size_t)row * Dn;
    const float* r_im = dim_ + (size_t)row * Dn;

    float vre[8], vim[8];
    float s_re = 0.f, s_im = 0.f;
    for (int i = 0; i < PER; i++) {
        int d = tid + i * 256;
        vre[i] = r_re[d];
        vim[i] = r_im[d];
        s_re += vre[i];
        s_im += vim[i];
    }
    __shared__ float red[2][256];
    red[0][tid] = s_re; red[1][tid] = s_im;
    __syncthreads();
    for (int off = 128; off > 0; off >>= 1) {
        if (tid < off) {
            red[0][tid] += red[0][tid + off];
            red[1][tid] += red[1][tid + off];
        }
        __syncthreads();
    }
    const float mu_re = red[0][0] / Dn;
    const float mu_im = red[1][0] / Dn;
    __syncthreads();

    float v = 0.f;
    for (int i = 0; i < PER; i++) {
        float cr = vre[i] - mu_re, ci = vim[i] - mu_im;
        v += cr * cr + ci * ci;
    }
    red[0][tid] = v;
    __syncthreads();
    for (int off = 128; off > 0; off >>= 1) {
        if (tid < off) red[0][tid] += red[0][tid + off];
        __syncthreads();
    }
    const float rstd = rsqrtf(red[0][0] / Dn + EPSL);

    for (int i = 0; i < PER; i++) {
        int d = tid + i * 256;
        float zr = (vre[i] - mu_re) * rstd;
        float zi = (vim[i] - mu_im) * rstd;
        float gr = gre[d], gi = gim[d];
        float o_re = gr * zr - gi * zi + bre[d] + xre[(size_t)row * Dn + d];
        float o_im = gr * zi + gi * zr + bim[d] + xim[(size_t)row * Dn + d];
        size_t o = ((size_t)row * Dn + d) * 2;
        out[o]     = o_re;
        out[o + 1] = o_im;
    }
}

extern "C" void kernel_launch(void* const* d_in, const int* in_sizes, int n_in,
                              void* d_out, int out_size, void* d_ws, size_t ws_size,
                              hipStream_t stream)
{
    const float* x_re    = (const float*)d_in[0];
    const float* x_im    = (const float*)d_in[1];
    const float* Wup_re  = (const float*)d_in[2];
    const float* Wup_im  = (const float*)d_in[3];
    const float* bias_re = (const float*)d_in[4];
    const float* bias_im = (const float*)d_in[5];
    const float* mb      = (const float*)d_in[6];
    const float* Wdn_re  = (const float*)d_in[7];
    const float* Wdn_im  = (const float*)d_in[8];
    const float* g_re    = (const float*)d_in[9];
    const float* g_im    = (const float*)d_in[10];
    const float* b_re    = (const float*)d_in[11];
    const float* b_im    = (const float*)d_in[12];

    const int Dd = in_sizes[9];       // D (ln_gamma_re)
    const int F  = in_sizes[6];       // F (modrelu_b)
    const int M  = in_sizes[0] / Dd;  // B*S

    // Workspace layout: up_re/up_im (bf16 [M][F]), down_re/down_im (f32 [M][D]).
    char* ws = (char*)d_ws;
    void* up_re = (void*)ws;
    void* up_im = (void*)(ws + (size_t)M * F * 2);
    void* dn_re = (void*)(ws + (size_t)M * F * 4);
    void* dn_im = (void*)(ws + (size_t)M * F * 4 + (size_t)M * Dd * 4);

    dim3 blk(256);

    dim3 g1(F / BN, M / BM);
    cgemm_kernel<false, 0><<<g1, blk, 0, stream>>>(
        x_re, x_im, Wup_re, Wup_im, bias_re, bias_im, mb, up_re, up_im, M, F, Dd);

    dim3 g2(Dd / BN, M / BM);
    cgemm_kernel<true, 1><<<g2, blk, 0, stream>>>(
        up_re, up_im, Wdn_re, Wdn_im, nullptr, nullptr, nullptr, dn_re, dn_im, M, Dd, F);

    ln_residual_kernel<<<dim3(M), blk, 0, stream>>>(
        (const float*)dn_re, (const float*)dn_im, x_re, x_im,
        g_re, g_im, b_re, b_im, (float*)d_out, Dd);
}
